// Net_81475529605882
// MI455X (gfx1250) — matrix operand order repaired and checked by reading, not compiled, not run on hardware
//
#include <hip/hip_runtime.h>

// ---------------------------------------------------------------------------
// MI455X (gfx1250) implementation, round 2.
// Heavy 64-channel convs -> implicit-GEMM with v_wmma_f32_16x16x32_bf16.
// Feature maps stored bf16 NHWC (channels innermost) so each K=32 WMMA step
// loads its A fragment as two contiguous 16B chunks per lane (b128 loads).
// Out-of-bounds taps redirect to a zero page (cndmask on pointer, no branchy
// per-element loads). Spatial<->angular permutes + feat5 transpose folded
// into strided addressing. Small disparity head stays fp32 VALU.
// ---------------------------------------------------------------------------

typedef __bf16 bf16_t;
typedef __attribute__((ext_vector_type(16))) __bf16 v16bf;
typedef __attribute__((ext_vector_type(8)))  __bf16 v8bf;
typedef __attribute__((ext_vector_type(8)))  float  v8f;

union AV { v16bf v; v8bf h[2]; };

#define AN   7
#define AN2  49
#define NS   4
#define Hh   96
#define Ww   96
#define HW   9216
#define VSTR (64 * HW)   // one 64-ch 96x96 frame (589824 elems)

// ---------------- generic direct fp32 conv2d (small layers) ----------------
__global__ void direct_conv2d(const float* __restrict__ in, const float* __restrict__ w,
                              const float* __restrict__ bias, void* __restrict__ out,
                              int out_bf16, int out_nhwc, int B, int Cin, int Cout,
                              int H, int W, int k, int pad, int dil, int relu)
{
    long total = (long)B * Cout * H * W;
    long idx = (long)blockIdx.x * blockDim.x + threadIdx.x;
    if (idx >= total) return;
    int x = (int)(idx % W); long t = idx / W;
    int y = (int)(t % H); t /= H;
    int oc = (int)(t % Cout); int b = (int)(t / Cout);
    float acc = bias[oc];
    const float* wp = w + (long)oc * Cin * k * k;
    const float* ip = in + (long)b * Cin * H * W;
    for (int c = 0; c < Cin; ++c) {
        for (int i = 0; i < k; ++i) {
            int iy = y + i * dil - pad;
            if (iy < 0 || iy >= H) continue;
            for (int j = 0; j < k; ++j) {
                int ix = x + j * dil - pad;
                if (ix < 0 || ix >= W) continue;
                acc = fmaf(ip[(long)c * H * W + (long)iy * W + ix],
                           wp[(c * k + i) * k + j], acc);
            }
        }
    }
    if (relu) acc = fmaxf(acc, 0.0f);
    long addr = out_nhwc ? ((long)b * Cout * H * W + ((long)y * W + x) * Cout + oc)
                         : idx;
    if (out_bf16) ((bf16_t*)out)[addr] = (bf16_t)acc;
    else          ((float*)out)[addr]  = acc;
}

// ---------------- bilinear warp (grid construction fused) -------------------
__global__ void warp_kernel(const float* __restrict__ img, const int* __restrict__ ind,
                            const float* __restrict__ disp, float* __restrict__ out)
{
    int idx = blockIdx.x * blockDim.x + threadIdx.x;
    if (idx >= AN2 * NS * HW) return;
    int pix = idx % HW; int t0 = idx / HW;
    int s = t0 % NS;   int t  = t0 / NS;
    int y = pix / Ww, x = pix % Ww;
    float d = disp[t * HW + pix];
    int it = ind[s];
    float shh = (float)(t / AN - it / AN);
    float shw = (float)(t % AN - it % AN);
    float gx = (float)x + d * shw;
    float gy = (float)y + d * shh;
    float x0f = floorf(gx), y0f = floorf(gy);
    float wx = gx - x0f, wy = gy - y0f;
    int x0 = (int)x0f, y0 = (int)y0f;
    const float* im = img + (long)s * HW;
    auto samp = [&](int yy, int xx) -> float {
        bool v = (xx >= 0) & (xx < Ww) & (yy >= 0) & (yy < Hh);
        int cy = min(max(yy, 0), Hh - 1);
        int cx = min(max(xx, 0), Ww - 1);
        float r = im[cy * Ww + cx];
        return v ? r : 0.0f;
    };
    float r = samp(y0, x0)         * (1.f - wx) * (1.f - wy)
            + samp(y0, x0 + 1)     * wx         * (1.f - wy)
            + samp(y0 + 1, x0)     * (1.f - wx) * wy
            + samp(y0 + 1, x0 + 1) * wx         * wy;
    out[((long)t * NS + s) * HW + pix] = r;
}

// --------- repack fp32 OIHW / OI(D)HW weights into per-lane B fragments ----
// Layout: Bpack[step][nblk][lane(N=lane&15)][e(K=8*(lane>>4)+(e&7)+16*(e>>3))]  (contiguous 32B/lane)
__global__ void repack_w(const float* __restrict__ w, bf16_t* __restrict__ out,
                         int Cout, int Cin, int kd, int NB, int nsteps)
{
    int total = nsteps * NB * 512;
    int idx = blockIdx.x * blockDim.x + threadIdx.x;
    if (idx >= total) return;
    int e    = idx & 15;
    int lane = (idx >> 4) & 31;
    int t    = idx >> 9;
    int nblk = t % NB; int s = t / NB;
    int tap = s >> 1; int cbase = (s & 1) << 5;
    int dz = tap / 9; int rr = tap % 9;
    int dy = rr / 3;  int dx = rr % 3;
    int oc = nblk * 16 + (lane & 15);
    int c  = cbase + ((lane >> 4) << 3) + (e & 7) + ((e >> 3) << 4);
    float v = 0.0f;
    if (oc < Cout)
        v = w[(((long)oc * Cin + c) * kd + dz) * 9 + dy * 3 + dx];
    out[idx] = (bf16_t)v;
}

// ---------------- zero page init ----------------
__global__ void zfill(bf16_t* z, int n)
{
    int i = blockIdx.x * blockDim.x + threadIdx.x;
    if (i < n) z[i] = (bf16_t)0.0f;
}

// ---------------- WMMA implicit-GEMM conv (2D & 3D, NHWC, strided) ---------
// One wave32 -> one 16(pixels) x 16(out-ch) tile. Channels contiguous, so a
// K=32 step loads A as two 16B chunks per lane. K walked as kd*9*2 steps.
__global__ void __launch_bounds__(256) wmma_conv(
    const bf16_t* __restrict__ in, const bf16_t* __restrict__ Bp,
    const float* __restrict__ bias, const bf16_t* __restrict__ zeros,
    void* __restrict__ out, int out_f32,
    int B, int H, int W,
    long in_sb, long in_sy, long in_sx, long in_sz, int zmul, int kd,
    long out_sb, long out_sc, long out_sp,
    int NB, int Cout, int relu)
{
    int HWp = H * W;
    int ptiles = (HWp + 15) >> 4;
    long ntiles = (long)B * ptiles * NB;
    long tile = (long)blockIdx.x * 8 + (threadIdx.x >> 5);
    if (tile >= ntiles) return;
    int lane = threadIdx.x & 31;
    int nblk = (int)(tile % NB); long t2 = tile / NB;
    int ptile = (int)(t2 % ptiles); int b = (int)(t2 / ptiles);

    int m = lane & 15;
    int pix = ptile * 16 + m;
    bool vm = pix < HWp;
    int y = pix / W, x = pix % W;
    int khalf = (lane >> 4) << 3;          // +8 K offset for lanes 16..31 (A)

    v8f accA = {0.f, 0.f, 0.f, 0.f, 0.f, 0.f, 0.f, 0.f};
    v8f accB = {0.f, 0.f, 0.f, 0.f, 0.f, 0.f, 0.f, 0.f};
    long bstep = (long)NB * 512;
    const bf16_t* bp = Bp + (long)nblk * 512 + (long)lane * 16;

    for (int dz = 0; dz < kd; ++dz) {
        long zbase = (long)b * in_sb + (long)(b * zmul + dz) * in_sz + khalf;
        const bf16_t* bz = bp + (long)(dz * 18) * bstep;
        #pragma unroll
        for (int t9 = 0; t9 < 9; ++t9) {
            const int dy = t9 / 3 - 1, dx = t9 % 3 - 1;
            int iy = y + dy, ix = x + dx;
            bool inb = vm && ((unsigned)iy < (unsigned)H) && ((unsigned)ix < (unsigned)W);
            const bf16_t* p = inb ? (in + zbase + (long)iy * in_sy + (long)ix * in_sx)
                                  : zeros;
            const bf16_t* bq = bz + (long)(t9 * 2) * bstep;
            // A fragments: ch-half 0 -> channels {khalf+0..7, khalf+16..23}
            //              ch-half 1 -> +32
            AV a0, a1;
            a0.h[0] = *(const v8bf*)(p);
            a0.h[1] = *(const v8bf*)(p + 16);
            a1.h[0] = *(const v8bf*)(p + 32);
            a1.h[1] = *(const v8bf*)(p + 48);
            v16bf b0 = *(const v16bf*)(bq);
            v16bf b1 = *(const v16bf*)(bq + bstep);
            __builtin_prefetch(bq + 2 * bstep, 0, 1);
            accA = __builtin_amdgcn_wmma_f32_16x16x32_bf16(
                false, a0.v, false, b0, (short)0, accA, false, false);
            accB = __builtin_amdgcn_wmma_f32_16x16x32_bf16(
                false, a1.v, false, b1, (short)0, accB, false, false);
        }
    }
    v8f acc = accA + accB;

    // Epilogue: C/D layout -> N = lane&15, M = reg + 8*(lane>=16)
    int n = lane & 15;
    int oc = nblk * 16 + n;
    float bs = (oc < Cout) ? bias[oc] : 0.0f;
    #pragma unroll
    for (int r = 0; r < 8; ++r) {
        int mo = r + khalf;
        int po = ptile * 16 + mo;
        if (po < HWp && oc < Cout) {
            float v = acc[r] + bs;
            if (relu) v = fmaxf(v, 0.0f);
            long addr = (long)b * out_sb + (long)oc * out_sc + (long)po * out_sp;
            if (out_f32) ((float*)out)[addr]  = v;
            else         ((bf16_t*)out)[addr] = (bf16_t)v;
        }
    }
}

// ---------------- final residual add: lf = warped[:,:,0] + r ----------------
__global__ void lf_kernel(const float* __restrict__ warped, const float* __restrict__ r3,
                          float* __restrict__ out)
{
    int idx = blockIdx.x * blockDim.x + threadIdx.x;
    if (idx >= AN2 * HW) return;
    int pix = idx % HW; int t = idx / HW;
    out[idx] = warped[(long)(t * NS) * HW + pix] + r3[idx];
}

// ---------------------------------------------------------------------------
extern "C" void kernel_launch(void* const* d_in, const int* in_sizes, int n_in,
                              void* d_out, int out_size, void* d_ws, size_t ws_size,
                              hipStream_t stream) {
    (void)in_sizes; (void)n_in; (void)out_size; (void)ws_size;
    auto F  = [&](int i) { return (const float*)d_in[i]; };
    const int* ind = (const int*)d_in[0];
    const float* img = F(1);

    // outputs: warped [49,4,96,96] | disp [49,96,96] | lf [49,96,96]
    float* out_warp = (float*)d_out;
    float* out_disp = out_warp + (size_t)AN2 * NS * HW;
    float* out_lf   = out_disp + (size_t)AN2 * HW;

    // workspace carve
    char* ws = (char*)d_ws;
    size_t off = 0;
    auto take = [&](size_t bytes) -> void* {
        size_t o = (off + 255) & ~(size_t)255;
        off = o + bytes;
        return (void*)(ws + o);
    };
    float*  dispA = (float*)take((size_t)64 * HW * 4);
    float*  dispB = (float*)take((size_t)64 * HW * 4);
    bf16_t* featA = (bf16_t*)take((size_t)AN2 * 64 * HW * 2);   // NHWC [view][pix][64]
    bf16_t* featB = (bf16_t*)take((size_t)AN2 * 64 * HW * 2);
    bf16_t* r1    = (bf16_t*)take((size_t)12 * 64 * HW * 2);    // NHWC [od][pix][64]
    bf16_t* r2    = (bf16_t*)take((size_t)3  * 64 * HW * 2);
    float*  r3    = (float*)take((size_t)AN2 * HW * 4);         // planar [oc][pix]
    bf16_t* zpage = (bf16_t*)take((size_t)256 * 2);
    bf16_t* pk_sp[4]; bf16_t* pk_an[4];
    for (int i = 0; i < 4; ++i) {
        pk_sp[i] = (bf16_t*)take((size_t)18 * 4 * 512 * 2);
        pk_an[i] = (bf16_t*)take((size_t)18 * 4 * 512 * 2);
    }
    bf16_t* pk_r1 = (bf16_t*)take((size_t)90 * 4 * 512 * 2);
    bf16_t* pk_r2 = (bf16_t*)take((size_t)72 * 4 * 512 * 2);
    bf16_t* pk_r3 = (bf16_t*)take((size_t)54 * 4 * 512 * 2);

    auto cdiv = [](long a, long b) { return (int)((a + b - 1) / b); };

    zfill<<<1, 256, 0, stream>>>(zpage, 256);

    // ---- disparity estimator (fp32 direct convs) ----
    struct { int cin, cout, k, pad, dil; } dc[9] = {
        {NS, 16, 7, 6, 2}, {16, 32, 7, 6, 2}, {32, 64, 5, 2, 1}, {64, 64, 5, 2, 1},
        {64, AN2, 3, 1, 1}, {AN2, AN2, 3, 1, 1}, {AN2, AN2, 3, 1, 1},
        {AN2, AN2, 3, 1, 1}, {AN2, AN2, 3, 1, 1}};
    const float* cur = img;
    for (int i = 0; i < 9; ++i) {
        void* dst = (i == 8) ? (void*)out_disp : ((i & 1) ? (void*)dispB : (void*)dispA);
        int relu = (i < 8);
        long elems = (long)dc[i].cout * HW;
        direct_conv2d<<<cdiv(elems, 256), 256, 0, stream>>>(
            cur, F(2 + 2 * i), F(3 + 2 * i), dst, /*bf16*/0, /*nhwc*/0,
            1, dc[i].cin, dc[i].cout, Hh, Ww, dc[i].k, dc[i].pad, dc[i].dil, relu);
        cur = (i & 1) ? dispB : dispA;
    }

    // ---- fused grid + bilinear warp ----
    warp_kernel<<<cdiv((long)AN2 * NS * HW, 256), 256, 0, stream>>>(
        img, ind, out_disp, out_warp);

    // ---- conv0: 4->64, fp32 in (warped NCHW), bf16 NHWC out ----
    direct_conv2d<<<cdiv((long)AN2 * 64 * HW, 256), 256, 0, stream>>>(
        out_warp, F(20), F(21), featA, /*bf16*/1, /*nhwc*/1,
        AN2, NS, 64, Hh, Ww, 3, 1, 1, /*relu*/1);

    // ---- weight repacks for WMMA convs ----
    for (int i = 0; i < 4; ++i) {
        repack_w<<<cdiv(18 * 4 * 512, 256), 256, 0, stream>>>(F(22 + 4 * i), pk_sp[i], 64, 64, 1, 4, 18);
        repack_w<<<cdiv(18 * 4 * 512, 256), 256, 0, stream>>>(F(24 + 4 * i), pk_an[i], 64, 64, 1, 4, 18);
    }
    repack_w<<<cdiv(90 * 4 * 512, 256), 256, 0, stream>>>(F(38), pk_r1, 64,  64, 5, 4, 90);
    repack_w<<<cdiv(72 * 4 * 512, 256), 256, 0, stream>>>(F(40), pk_r2, 64,  64, 4, 4, 72);
    repack_w<<<cdiv(54 * 4 * 512, 256), 256, 0, stream>>>(F(42), pk_r3, AN2, 64, 3, 4, 54);

    // ---- 4 alternating spatial/angular WMMA convs (NHWC) ----
    for (int i = 0; i < 4; ++i) {
        // spatial: element (view b, c, y, x) -> b*VSTR + (y*96+x)*64 + c
        long nt_sp = (long)AN2 * ((HW + 15) / 16) * 4;
        wmma_conv<<<cdiv(nt_sp, 8), 256, 0, stream>>>(
            featA, pk_sp[i], F(23 + 4 * i), zpage, featB, /*f32*/0,
            AN2, Hh, Ww,
            /*in_sb*/VSTR, /*in_sy*/(long)Ww * 64, /*in_sx*/64, /*in_sz*/0, 0, /*kd*/1,
            /*out_sb*/VSTR, /*out_sc*/1, /*out_sp*/64,
            /*NB*/4, /*Cout*/64, /*relu*/1);
        // angular: element (pix b, c, ay, ax) -> (ay*7+ax)*VSTR + b*64 + c
        long nt_an = (long)HW * ((AN2 + 15) / 16) * 4;
        wmma_conv<<<cdiv(nt_an, 8), 256, 0, stream>>>(
            featB, pk_an[i], F(25 + 4 * i), zpage, featA, /*f32*/0,
            HW, AN, AN,
            /*in_sb*/64, /*in_sy*/(long)AN * VSTR, /*in_sx*/VSTR, /*in_sz*/0, 0, 1,
            /*out_sb*/64, /*out_sc*/1, /*out_sp*/VSTR,
            4, 64, 1);
    }

    // ---- 3D residual convs (feat5 transpose folded into z-strides) ----
    long nt1 = (long)12 * ((HW + 15) / 16) * 4;
    wmma_conv<<<cdiv(nt1, 8), 256, 0, stream>>>(
        featA, pk_r1, F(39), zpage, r1, 0,
        12, Hh, Ww,
        /*in_sb*/0, /*in_sy*/(long)Ww * 64, /*in_sx*/64, /*in_sz*/VSTR, /*zmul*/4, /*kd*/5,
        /*out_sb*/VSTR, /*out_sc*/1, /*out_sp*/64,
        4, 64, 1);
    long nt2 = (long)3 * ((HW + 15) / 16) * 4;
    wmma_conv<<<cdiv(nt2, 8), 256, 0, stream>>>(
        r1, pk_r2, F(41), zpage, r2, 0,
        3, Hh, Ww,
        0, (long)Ww * 64, 64, VSTR, 4, /*kd*/4,
        VSTR, 1, 64,
        4, 64, 1);
    long nt3 = (long)1 * ((HW + 15) / 16) * 4;
    wmma_conv<<<cdiv(nt3, 8), 256, 0, stream>>>(
        r2, pk_r3, F(43), zpage, r3, /*f32*/1,
        1, Hh, Ww,
        0, (long)Ww * 64, 64, VSTR, 1, /*kd*/3,
        /*out_sb*/0, /*out_sc*/HW, /*out_sp*/1,
        4, AN2, /*relu*/0);

    // ---- lf = warped[:,:,0] + r ----
    lf_kernel<<<cdiv((long)AN2 * HW, 256), 256, 0, stream>>>(out_warp, r3, out_lf);
}
